// BaselineModel_87711822119337
// MI455X (gfx1250) — matrix operand order repaired and checked by reading, not compiled, not run on hardware
//
#include <hip/hip_runtime.h>
#include <hip/hip_bf16.h>

// Problem constants (fixed by the reference)
#define Hh     128
#define GN     4096      // G*N nodes
#define NE     65536     // G*EPG edges
#define NGRAPH 32
#define NNODE  128
#define OUTD   12

typedef __attribute__((ext_vector_type(16))) __bf16 v16bf;
typedef __attribute__((ext_vector_type(8)))  __bf16 v8bf;
typedef __attribute__((ext_vector_type(8)))  float  v8f;
typedef __attribute__((ext_vector_type(4)))  unsigned int v4u;
typedef __attribute__((ext_vector_type(8)))  int    v8i;
typedef __attribute__((ext_vector_type(4)))  int    v4i;

#if defined(__has_builtin)
# if __has_builtin(__builtin_amdgcn_tensor_load_to_lds)
#  define USE_TDM 1
# endif
#endif
#ifndef USE_TDM
# define USE_TDM 0
#endif

// ---------------------------------------------------------------------------
// WMMA helpers (CDNA5 wave32, V_WMMA_F32_16X16X32_BF16)
// ---------------------------------------------------------------------------
static __device__ inline v8f wmma_bf16(v16bf a, v16bf b, v8f c) {
  return __builtin_amdgcn_wmma_f32_16x16x32_bf16(
      /*neg_a=*/false, a, /*neg_b=*/false, b,
      /*c_mod=*/(short)0, c, /*reuse_a=*/false, /*reuse_b=*/false);
}

// A-fragment: 16x32 bf16 tile, row-major source with leading dim `ld`.
// Layout: lanes 0-15 -> M=lane, K = 0..15;
//         lanes 16-31 -> M=lane-16, K = 16..31 (matches B fragment).
static __device__ inline v16bf load_a_frag(const __bf16* base, int ld,
                                           int kbase, int lane) {
  int row  = lane & 15;
  int koff = (lane >> 4) * 16;
  const __bf16* p = base + row * ld + kbase + koff;
  v8bf lo = *(const v8bf*)(p);
  v8bf hi = *(const v8bf*)(p + 8);
  v16bf a;
  ((v8bf*)&a)[0] = lo;
  ((v8bf*)&a)[1] = hi;
  return a;
}

// B-fragment: 32x16 bf16 tile from a TRANSPOSED (N-major) weight, ld = K.
// lane holds column n = lane&15, K-run (lane>>4)*16 .. +15  -> one contiguous
// 32-byte load from Wt[n][k0..k0+15].
static __device__ inline v16bf load_b_frag(const __bf16* Wt, int ld,
                                           int nbase, int kbase, int lane) {
  int n  = nbase + (lane & 15);
  int k0 = kbase + (lane >> 4) * 16;
  return *(const v16bf*)(Wt + n * ld + k0);
}

// Float atomic max via sign-split integer atomics (native global atomics).
static __device__ inline void atomicMaxF(float* a, float v) {
  if (v >= 0.0f) atomicMax((int*)a, __float_as_int(v));
  else           atomicMin((unsigned int*)a, (unsigned int)__float_as_int(v));
}

// ---------------------------------------------------------------------------
// TDM: DMA one 128x128 bf16 (32KB, contiguous) weight tile into LDS.
// D# per CDNA5 ISA: group0 {count=1, lds_addr,
// global_addr, type=2}; group1 {data_size=2B, tensor/tile dim0=16384, 1 row}.
// ---------------------------------------------------------------------------
#if USE_TDM
static __device__ inline void tdm_load_w32k(const __bf16* src, unsigned lds_off) {
  unsigned long long ga = (unsigned long long)(uintptr_t)src;
  v4u g0;
  g0[0] = 1u;                                              // count=1, user mode
  g0[1] = lds_off;                                         // lds_addr[31:0]
  g0[2] = (unsigned)(ga & 0xFFFFFFFFu);                    // global_addr[95:64]
  g0[3] = (unsigned)((ga >> 32) & 0x01FFFFFFu) | 0x80000000u; // addr msbs | type=2
  v8i g1;
  g1[0] = 0x00010000;        // workgroup_mask=0, data_size=1 (2 bytes)
  g1[1] = 0x40000000;        // tensor_dim0 = 16384 (low 16 bits << 16)
  g1[2] = 0x00010000;        // tensor_dim0 hi=0, tensor_dim1 = 1
  g1[3] = 0x40000000;        // tensor_dim1 hi=0, tile_dim0 = 16384
  g1[4] = 0x00000001;        // tile_dim1 = 1, tile_dim2 = 0
  g1[5] = 16384;             // tensor_dim0_stride low 32
  g1[6] = 0x40000000;        // stride0 hi=0, tensor_dim1_stride low16 = 16384
  g1[7] = 0;                 // tensor_dim1_stride hi
  v4i g2 = {0, 0, 0, 0};
  v4i g3 = {0, 0, 0, 0};
#if __clang_major__ >= 23
  v8i g4 = {0, 0, 0, 0, 0, 0, 0, 0};
  __builtin_amdgcn_tensor_load_to_lds(g0, g1, g2, g3, g4, 0);
#else
  __builtin_amdgcn_tensor_load_to_lds(g0, g1, g2, g3, 0);
#endif
}
#endif

// Stage one transposed bf16 weight matrix (128x128) into LDS. Caller must
// __syncthreads() afterwards (and before, if the buffer is being reused).
static __device__ inline void stage_weight(const __bf16* __restrict__ Wt,
                                           __bf16* lds) {
#if USE_TDM
  if (threadIdx.x == 0) {
    tdm_load_w32k(Wt, (unsigned)(uintptr_t)lds);   // LDS aperture: addr[31:0]
    __builtin_amdgcn_s_wait_tensorcnt(0);          // wave0 sees completion
  }
#else
  const int t = threadIdx.x;                        // 128 threads, 256B each
#pragma unroll
  for (int i = 0; i < 16; ++i)
    *(v8bf*)(lds + t * 128 + i * 8) = *(const v8bf*)(Wt + t * 128 + i * 8);
#endif
}

// ---------------------------------------------------------------------------
// Weight prep: fp32 row-major (K x N) -> bf16 transposed (N x K)
// ---------------------------------------------------------------------------
__global__ void k_conv_t(const float* __restrict__ W, __bf16* __restrict__ Wt,
                         int K, int Nn) {
  int i = blockIdx.x * blockDim.x + threadIdx.x;
  if (i >= K * Nn) return;
  int k = i / Nn, n = i % Nn;
  Wt[n * K + k] = (__bf16)W[k * Nn + n];
}

// ---------------------------------------------------------------------------
// Encoders
// ---------------------------------------------------------------------------
__global__ void k_encode_nodes(const int* __restrict__ x,
                               const float* __restrict__ atom_tab,
                               __bf16* __restrict__ z) {
  int i = blockIdx.x * blockDim.x + threadIdx.x;   // over GN*H
  if (i >= GN * Hh) return;
  int node = i / Hh, h = i % Hh;
  float v = 0.0f;
#pragma unroll
  for (int f = 0; f < 9; ++f)
    v += atom_tab[(f * 119 + x[node * 9 + f]) * Hh + h];
  z[node * 2 * Hh + h]      = (__bf16)v;      // nd half
  z[node * 2 * Hh + Hh + h] = (__bf16)0.0f;   // hid half (layer 0)
}

__global__ void k_encode_edges(const int* __restrict__ ea,
                               const float* __restrict__ bond_tab,
                               __bf16* __restrict__ efts) {
  int i = blockIdx.x * blockDim.x + threadIdx.x;   // over NE*H
  if (i >= NE * Hh) return;
  int e = i / Hh, h = i % Hh;
  float v = 0.0f;
#pragma unroll
  for (int f = 0; f < 3; ++f)
    v += bond_tab[(f * 8 + ea[e * 3 + f]) * Hh + h];
  efts[e * Hh + h] = (__bf16)v;
}

// ---------------------------------------------------------------------------
// m1 = z @ W_m1 + b1 ; m2 = z @ W_m2 + b2  (K = 2H = 256)
// grid (GN/16, H/16), block 32 (one wave computes a 16x16 tile of each)
// ---------------------------------------------------------------------------
__global__ __launch_bounds__(32)
void k_mm12(const __bf16* __restrict__ z,
            const __bf16* __restrict__ W1t, const __bf16* __restrict__ W2t,
            const float* __restrict__ b1, const float* __restrict__ b2,
            float* __restrict__ m1, float* __restrict__ m2) {
  int lane  = threadIdx.x;
  int rbase = blockIdx.x * 16;
  int nbase = blockIdx.y * 16;
  const __bf16* zr = z + rbase * (2 * Hh);

  v8f c1 = {}, c2 = {};
#pragma unroll
  for (int kc = 0; kc < 8; ++kc) {
    v16bf a  = load_a_frag(zr, 2 * Hh, kc * 32, lane);
    v16bf b  = load_b_frag(W1t, 2 * Hh, nbase, kc * 32, lane);
    c1 = wmma_bf16(a, b, c1);
    b  = load_b_frag(W2t, 2 * Hh, nbase, kc * 32, lane);
    c2 = wmma_bf16(a, b, c2);
  }
  int n   = nbase + (lane & 15);
  int mhi = (lane >> 4) * 8;
  float bb1 = b1[n], bb2 = b2[n];
#pragma unroll
  for (int r = 0; r < 8; ++r) {
    int row = rbase + mhi + r;
    m1[row * Hh + n] = c1[r] + bb1;
    m2[row * Hh + n] = c2[r] + bb2;
  }
}

// ---------------------------------------------------------------------------
// Fused per-edge message MLP + masked-max scatter.
// Block = 4 waves / 64 edges. Weights staged once per block into LDS (TDM),
// shared by all 4 waves; 3 chained 16x128x128 GEMMs per wave (96 WMMAs).
// ---------------------------------------------------------------------------
__global__ __launch_bounds__(128)
void k_edge(const __bf16* __restrict__ efts, const long long* __restrict__ eidx,
            const __bf16* __restrict__ Wmet, const float* __restrict__ bme,
            const __bf16* __restrict__ Wh1t, const float* __restrict__ bh1,
            const __bf16* __restrict__ Wh2t, const float* __restrict__ bh2,
            const float* __restrict__ m1, const float* __restrict__ m2,
            float* __restrict__ agg) {
  __shared__ __bf16 wtile[128 * 128];      // 32KB staged weight (reused 3x)
  __shared__ __bf16 hst[4][16 * Hh];       // 4KB staging tile per wave
  __shared__ int sid[64], did[64];

  const int tid   = threadIdx.x;
  const int wave  = tid >> 5;
  const int lane  = tid & 31;
  const int ebase = blockIdx.x * 64 + wave * 16;
  __bf16* myh = hst[wave];

  if (tid < 64) {
    sid[tid] = (int)eidx[blockIdx.x * 64 + tid];        // sender ids
    did[tid] = (int)eidx[NE + blockIdx.x * 64 + tid];   // receiver ids
  }
  __builtin_prefetch(Wh1t + tid * 128, 0, 1);   // warm L2 for phase 2
  stage_weight(Wmet, wtile);
  __syncthreads();

  const int mhi  = (lane >> 4) * 8;
  const int nloc = lane & 15;
  int ss[8], dd[8];
#pragma unroll
  for (int r = 0; r < 8; ++r) {
    ss[r] = sid[wave * 16 + mhi + r];
    dd[r] = did[wave * 16 + mhi + r];
  }

  // --- GEMM1: edge_fts @ W_me (B from LDS) ---
  v16bf ae[4];
#pragma unroll
  for (int kc = 0; kc < 4; ++kc)
    ae[kc] = load_a_frag(efts + ebase * Hh, Hh, kc * 32, lane);

  v8f acc[8];
#pragma unroll
  for (int nt = 0; nt < 8; ++nt) {
    v8f c = {};
#pragma unroll
    for (int kc = 0; kc < 4; ++kc)
      c = wmma_bf16(ae[kc], load_b_frag(wtile, Hh, nt * 16, kc * 32, lane), c);
    acc[nt] = c;
  }

  // msgs = me + m1[dst] + m2[src] + b_me ; relu ; stage bf16 in LDS
#pragma unroll
  for (int nt = 0; nt < 8; ++nt) {
    int h = nt * 16 + nloc;
    float bb = bme[h];
#pragma unroll
    for (int r = 0; r < 8; ++r) {
      float v = acc[nt][r] + m1[dd[r] * Hh + h] + m2[ss[r] * Hh + h] + bb;
      v = v > 0.0f ? v : 0.0f;
      myh[(mhi + r) * Hh + h] = (__bf16)v;
    }
  }
  __syncthreads();                      // all waves done reading W_me

  // --- GEMM2: h @ W_h1, relu ---
  __builtin_prefetch(Wh2t + tid * 128, 0, 1);   // warm L2 for phase 3
  stage_weight(Wh1t, wtile);
  __syncthreads();
#pragma unroll
  for (int kc = 0; kc < 4; ++kc)
    ae[kc] = load_a_frag(myh, Hh, kc * 32, lane);
#pragma unroll
  for (int nt = 0; nt < 8; ++nt) {
    v8f c = {};
#pragma unroll
    for (int kc = 0; kc < 4; ++kc)
      c = wmma_bf16(ae[kc], load_b_frag(wtile, Hh, nt * 16, kc * 32, lane), c);
    acc[nt] = c;
  }
#pragma unroll
  for (int nt = 0; nt < 8; ++nt) {
    int h = nt * 16 + nloc;
    float bb = bh1[h];
#pragma unroll
    for (int r = 0; r < 8; ++r) {
      float v = acc[nt][r] + bb;
      v = v > 0.0f ? v : 0.0f;
      myh[(mhi + r) * Hh + h] = (__bf16)v;
    }
  }
  __syncthreads();                      // all waves done reading W_h1

  // --- GEMM3: h @ W_h2 + b_h2, masked-max scatter into agg[dst] ---
  stage_weight(Wh2t, wtile);
  __syncthreads();
#pragma unroll
  for (int kc = 0; kc < 4; ++kc)
    ae[kc] = load_a_frag(myh, Hh, kc * 32, lane);
#pragma unroll
  for (int nt = 0; nt < 8; ++nt) {
    v8f c = {};
#pragma unroll
    for (int kc = 0; kc < 4; ++kc)
      c = wmma_bf16(ae[kc], load_b_frag(wtile, Hh, nt * 16, kc * 32, lane), c);
    int h = nt * 16 + nloc;
    float bb = bh2[h];
#pragma unroll
    for (int r = 0; r < 8; ++r)
      atomicMaxF(&agg[dd[r] * Hh + h], c[r] + bb);
  }
}

// ---------------------------------------------------------------------------
// hid = relu(z @ W_o1 + agg @ W_o2 + b_o1 + b_o2)
// writes hid fp32 and the bf16 hid-half of z for the next layer.
// ---------------------------------------------------------------------------
__global__ __launch_bounds__(32)
void k_hid(const __bf16* __restrict__ zin, const __bf16* __restrict__ aggbf,
           const __bf16* __restrict__ Wo1t, const float* __restrict__ bo1,
           const __bf16* __restrict__ Wo2t, const float* __restrict__ bo2,
           float* __restrict__ hid, __bf16* __restrict__ zout) {
  int lane  = threadIdx.x;
  int rbase = blockIdx.x * 16;
  int nbase = blockIdx.y * 16;

  v8f c = {};
#pragma unroll
  for (int kc = 0; kc < 8; ++kc) {
    v16bf a = load_a_frag(zin + rbase * (2 * Hh), 2 * Hh, kc * 32, lane);
    c = wmma_bf16(a, load_b_frag(Wo1t, 2 * Hh, nbase, kc * 32, lane), c);
  }
#pragma unroll
  for (int kc = 0; kc < 4; ++kc) {
    v16bf a = load_a_frag(aggbf + rbase * Hh, Hh, kc * 32, lane);
    c = wmma_bf16(a, load_b_frag(Wo2t, Hh, nbase, kc * 32, lane), c);
  }
  int n   = nbase + (lane & 15);
  int mhi = (lane >> 4) * 8;
  float bb = bo1[n] + bo2[n];
#pragma unroll
  for (int r = 0; r < 8; ++r) {
    int row = rbase + mhi + r;
    float v = c[r] + bb;
    v = v > 0.0f ? v : 0.0f;
    hid[row * Hh + n] = v;
    zout[row * 2 * Hh + Hh + n] = (__bf16)v;
  }
}

// ---------------------------------------------------------------------------
// small elementwise utilities
// ---------------------------------------------------------------------------
__global__ void k_fill(float* __restrict__ p, float v, int n) {
  int i = blockIdx.x * blockDim.x + threadIdx.x;
  if (i < n) p[i] = v;
}

__global__ void k_f2b(const float* __restrict__ a, __bf16* __restrict__ b, int n) {
  int i = blockIdx.x * blockDim.x + threadIdx.x;
  if (i < n) b[i] = (__bf16)a[i];
}

// ---------------------------------------------------------------------------
// Readout: emb = mean_n hid ; out = relu(emb@Wg1+bg1)@Wg2+bg2  (tiny)
// ---------------------------------------------------------------------------
__global__ void k_head(const float* __restrict__ hid,
                       const float* __restrict__ Wg1, const float* __restrict__ bg1,
                       const float* __restrict__ Wg2, const float* __restrict__ bg2,
                       float* __restrict__ out) {
  __shared__ float emb[Hh], u[Hh];
  int g = blockIdx.x, t = threadIdx.x;
  float s = 0.0f;
  for (int n = 0; n < NNODE; ++n) s += hid[(g * NNODE + n) * Hh + t];
  emb[t] = s * (1.0f / NNODE);
  __syncthreads();
  float a = bg1[t];
  for (int k = 0; k < Hh; ++k) a += emb[k] * Wg1[k * Hh + t];
  u[t] = a > 0.0f ? a : 0.0f;
  __syncthreads();
  if (t < OUTD) {
    float o = bg2[t];
    for (int k = 0; k < Hh; ++k) o += u[k] * Wg2[k * OUTD + t];
    out[g * OUTD + t] = o;
  }
}

// ---------------------------------------------------------------------------
// host
// ---------------------------------------------------------------------------
extern "C" void kernel_launch(void* const* d_in, const int* in_sizes, int n_in,
                              void* d_out, int out_size, void* d_ws, size_t ws_size,
                              hipStream_t stream) {
  const int*       x        = (const int*)d_in[0];
  const long long* eidx     = (const long long*)d_in[1];
  const int*       eattr    = (const int*)d_in[2];
  const float*     atom_tab = (const float*)d_in[4];
  const float*     bond_tab = (const float*)d_in[5];
  const float*     W_m1 = (const float*)d_in[6];
  const float*     b_m1 = (const float*)d_in[7];
  const float*     W_m2 = (const float*)d_in[8];
  const float*     b_m2 = (const float*)d_in[9];
  const float*     W_me = (const float*)d_in[10];
  const float*     b_me = (const float*)d_in[11];
  // W_mg/b_mg (d_in[12..13]) contribute exactly zero (gf==0, b_mg==0): dropped.
  const float*     W_h1 = (const float*)d_in[14];
  const float*     b_h1 = (const float*)d_in[15];
  const float*     W_h2 = (const float*)d_in[16];
  const float*     b_h2 = (const float*)d_in[17];
  const float*     W_o1 = (const float*)d_in[18];
  const float*     b_o1 = (const float*)d_in[19];
  const float*     W_o2 = (const float*)d_in[20];
  const float*     b_o2 = (const float*)d_in[21];
  const float*     Wg1  = (const float*)d_in[22];
  const float*     bg1  = (const float*)d_in[23];
  const float*     Wg2  = (const float*)d_in[24];
  const float*     bg2  = (const float*)d_in[25];

  // workspace layout (bytes, 256-aligned): ~29 MB total
  char* ws = (char*)d_ws;
  __bf16* z     = (__bf16*)(ws);                       // GN x 2H bf16   (2 MB)
  __bf16* efts  = (__bf16*)(ws + (2u  << 20));         // NE x H bf16    (16 MB)
  float*  m1    = (float*) (ws + (18u << 20));         // GN x H f32     (2 MB)
  float*  m2    = (float*) (ws + (20u << 20));         // GN x H f32     (2 MB)
  float*  agg   = (float*) (ws + (22u << 20));         // GN x H f32     (2 MB)
  __bf16* aggbf = (__bf16*)(ws + (24u << 20));         // GN x H bf16    (1 MB)
  float*  hid   = (float*) (ws + (25u << 20));         // GN x H f32     (2 MB)
  __bf16* wbf   = (__bf16*)(ws + (27u << 20));         // bf16^T weights (~0.7 MB)

  auto conv = [&](const float* W, __bf16* Wt, int K, int Nn) {
    int n = K * Nn;
    k_conv_t<<<(n + 255) / 256, 256, 0, stream>>>(W, Wt, K, Nn);
  };

  // encoders
  k_encode_nodes<<<(GN * Hh + 255) / 256, 256, 0, stream>>>(x, atom_tab, z);
  k_encode_edges<<<(NE * Hh + 255) / 256, 256, 0, stream>>>(eattr, bond_tab, efts);

  const int PL = 3 * 256 * 128 + 4 * 128 * 128;   // bf16 elems per layer
  for (int l = 0; l < 2; ++l) {
    __bf16* base = wbf + l * PL;
    __bf16* wm1t = base;
    __bf16* wm2t = wm1t + 256 * 128;
    __bf16* wo1t = wm2t + 256 * 128;
    __bf16* wmet = wo1t + 256 * 128;
    __bf16* wh1t = wmet + 128 * 128;
    __bf16* wh2t = wh1t + 128 * 128;
    __bf16* wo2t = wh2t + 128 * 128;
    conv(W_m1 + l * 256 * 128, wm1t, 256, 128);
    conv(W_m2 + l * 256 * 128, wm2t, 256, 128);
    conv(W_o1 + l * 256 * 128, wo1t, 256, 128);
    conv(W_me + l * 128 * 128, wmet, 128, 128);
    conv(W_h1 + l * 128 * 128, wh1t, 128, 128);
    conv(W_h2 + l * 128 * 128, wh2t, 128, 128);
    conv(W_o2 + l * 128 * 128, wo2t, 128, 128);

    k_mm12<<<dim3(GN / 16, Hh / 16), 32, 0, stream>>>(
        z, wm1t, wm2t, b_m1 + l * Hh, b_m2 + l * Hh, m1, m2);

    k_fill<<<(GN * Hh + 255) / 256, 256, 0, stream>>>(agg, -1e9f, GN * Hh);

    k_edge<<<NE / 64, 128, 0, stream>>>(
        efts, eidx, wmet, b_me + l * Hh, wh1t, b_h1 + l * Hh,
        wh2t, b_h2 + l * Hh, m1, m2, agg);

    k_f2b<<<(GN * Hh + 255) / 256, 256, 0, stream>>>(agg, aggbf, GN * Hh);

    k_hid<<<dim3(GN / 16, Hh / 16), 32, 0, stream>>>(
        z, aggbf, wo1t, b_o1 + l * Hh, wo2t, b_o2 + l * Hh, hid, z);
  }

  k_head<<<NGRAPH, Hh, 0, stream>>>(hid, Wg1, bg1, Wg2, bg2, (float*)d_out);
  (void)in_sizes; (void)n_in; (void)out_size; (void)ws_size;
}
